// SelfAttention_70746701300418
// MI455X (gfx1250) — matrix operand order repaired and checked by reading, not compiled, not run on hardware
//
#include <hip/hip_runtime.h>
#include <hip/hip_bf16.h>
#include <stdint.h>

typedef __attribute__((ext_vector_type(16))) _Float16 v16h;
typedef __attribute__((ext_vector_type(8)))  _Float16 v8h;
typedef __attribute__((ext_vector_type(8)))  float    v8f;
typedef __attribute__((ext_vector_type(4)))  int      v4i;

#define LDSS 40   // f16 elements per LDS row: 32 data + 8 pad (80B = 20 dwords, conflict-free, 16B-aligned)

// gfx1250 async global->LDS copy (ASYNCcnt-tracked), guarded so the file compiles either way.
#if defined(__HIP_DEVICE_COMPILE__) && __has_builtin(__builtin_amdgcn_global_load_async_to_lds_b128)
#define HAS_ASYNC_LDS 1
typedef __attribute__((address_space(1))) v4i g_v4i;   // global int4
typedef __attribute__((address_space(3))) v4i l_v4i;   // LDS int4
#define ASYNC_CP16(ldsp, gp) \
  __builtin_amdgcn_global_load_async_to_lds_b128((g_v4i*)(gp), (l_v4i*)(ldsp), 0, 0)
#if __has_builtin(__builtin_amdgcn_s_wait_asynccnt)
#define AWAIT0() __builtin_amdgcn_s_wait_asynccnt(0)
#else
#define AWAIT0() asm volatile("s_wait_asynccnt 0x0" ::: "memory")
#endif
#else
#define HAS_ASYNC_LDS 0
#define AWAIT0()
#endif

// MODE 0: Q projection  (A=f32 x, B=f32 W, out f16 row-major, +bias, *scale)
// MODE 1: K projection  (same, no scale)
// MODE 2: V projection  (same, out stored transposed per-batch: [b][d][n])
// MODE 3: scores QK^T   (A=f16 Q, B=f16 K, causal mask, out f16; fully-masked tiles skipped)
// MODE 4: PV            (A=f16 P, B=f16 V^T, out f32; K clamped to bm+128 by causality)
template<int MODE>
__global__ __launch_bounds__(256) void wmma_gemm(
    const void* __restrict__ Ag, const void* __restrict__ Bg,
    void* __restrict__ outp, const float* __restrict__ bias,
    float scale, int M, int N, int K, int lda, int ldb)
{
  constexpr bool AB32 = (MODE <= 2);
  constexpr int  NBUF = (!AB32 && HAS_ASYNC_LDS) ? 2 : 1;
  __shared__ _Float16 lA[NBUF][128 * LDSS];
  __shared__ _Float16 lB[NBUF][128 * LDSS];

  const int t    = threadIdx.x;
  const int bm   = blockIdx.x * 128;
  const int bn   = blockIdx.y * 128;
  const int wave = t >> 5;
  const int lane = t & 31;
  const int wm   = (wave >> 1) * 32;   // wave's M offset in block tile
  const int wn   = (wave & 1) * 64;    // wave's N offset in block tile
  const int lrow = lane & 15;
  const int half = lane >> 4;

  const int ldr = t >> 1;              // loader row 0..127
  const int ldc = (t & 1) * 16;        // loader col group 0/16
  const int lofs = ldr * LDSS + ldc;   // loader LDS offset (f16 units)

  // ---- causal fast path: tile entirely above the diagonal -> pure mask store ----
  if constexpr (MODE == 3) {
    if (bn >= bm + 128) {
      v8h mval;
#pragma unroll
      for (int i = 0; i < 8; ++i) mval[i] = (_Float16)(-30000.0f);
      _Float16* dst = (_Float16*)outp + (size_t)(bm + ldr) * N + bn + (t & 1) * 64;
#pragma unroll
      for (int i = 0; i < 8; ++i) *(v8h*)(dst + 8 * i) = mval;
      return;
    }
  }

  int Kend = K;
  if constexpr (MODE == 4) Kend = (bm + 128 < K) ? bm + 128 : K;  // P rows are 0 past diagonal
  const int nk = Kend >> 5;

  v8f acc[2][4] = {};

  // ---- per-k-tile WMMA compute from an LDS buffer pair ----
  auto compute_tile = [&](const _Float16* __restrict__ Ab, const _Float16* __restrict__ Bb) {
    v16h aF[2], bF[4];
#pragma unroll
    for (int mi = 0; mi < 2; ++mi) {
      const _Float16* base = Ab + (wm + mi * 16 + lrow) * LDSS;
      v8h lo = *(const v8h*)(base + 16 * half);
      v8h hi = *(const v8h*)(base + 16 * half + 8);
      aF[mi] = __builtin_shufflevector(lo, hi, 0,1,2,3,4,5,6,7,8,9,10,11,12,13,14,15);
    }
#pragma unroll
    for (int ni = 0; ni < 4; ++ni) {
      const _Float16* base = Bb + (wn + ni * 16 + lrow) * LDSS + 16 * half;
      v8h lo = *(const v8h*)(base);
      v8h hi = *(const v8h*)(base + 8);
      bF[ni] = __builtin_shufflevector(lo, hi, 0,1,2,3,4,5,6,7,8,9,10,11,12,13,14,15);
    }
#pragma unroll
    for (int mi = 0; mi < 2; ++mi)
#pragma unroll
      for (int ni = 0; ni < 4; ++ni)
        acc[mi][ni] = __builtin_amdgcn_wmma_f32_16x16x32_f16(
            false, aF[mi], false, bF[ni], (short)0, acc[mi][ni], false, false);
  };

  if constexpr (AB32) {
    // f32 inputs: register-prefetch next tile's global loads, convert+store each iter.
    const float* pa0 = (const float*)Ag + (size_t)(bm + ldr) * lda + ldc;
    const float* pb0 = (const float*)Bg + (size_t)(bn + ldr) * ldb + ldc;
    float4 ra[4], rb[4];
    auto gload = [&](int kb) {
#pragma unroll
      for (int i = 0; i < 4; ++i) {
        ra[i] = *(const float4*)(pa0 + kb + 4 * i);
        rb[i] = *(const float4*)(pb0 + kb + 4 * i);
      }
    };
    auto cstore = [&]() {
      _Float16* da = &lA[0][lofs];
      _Float16* db = &lB[0][lofs];
#pragma unroll
      for (int i = 0; i < 4; ++i) {
        da[4*i+0] = (_Float16)ra[i].x; da[4*i+1] = (_Float16)ra[i].y;
        da[4*i+2] = (_Float16)ra[i].z; da[4*i+3] = (_Float16)ra[i].w;
        db[4*i+0] = (_Float16)rb[i].x; db[4*i+1] = (_Float16)rb[i].y;
        db[4*i+2] = (_Float16)rb[i].z; db[4*i+3] = (_Float16)rb[i].w;
      }
    };
    gload(0);
    for (int tk = 0; tk < nk; ++tk) {
      cstore();
      __syncthreads();
      if (tk + 1 < nk) gload((tk + 1) * 32);   // overlap with compute
      compute_tile(lA[0], lB[0]);
      __syncthreads();
    }
  } else {
    const _Float16* pa0 = (const _Float16*)Ag + (size_t)(bm + ldr) * lda + ldc;
    const _Float16* pb0 = (const _Float16*)Bg + (size_t)(bn + ldr) * ldb + ldc;
#if HAS_ASYNC_LDS
    // Double-buffered: async-fill buf[(t+1)&1] while WMMAs consume buf[t&1].
    auto afill = [&](int buf, int kb) {
      ASYNC_CP16(&lA[buf][lofs],     pa0 + kb);
      ASYNC_CP16(&lA[buf][lofs + 8], pa0 + kb + 8);
      ASYNC_CP16(&lB[buf][lofs],     pb0 + kb);
      ASYNC_CP16(&lB[buf][lofs + 8], pb0 + kb + 8);
    };
    afill(0, 0);
    AWAIT0();
    __syncthreads();
    for (int tk = 0; tk < nk; ++tk) {
      if (tk + 1 < nk) afill((tk + 1) & 1, (tk + 1) * 32);
      compute_tile(lA[tk & 1], lB[tk & 1]);
      AWAIT0();          // next buffer's fill (issued above) has completed
      __syncthreads();
    }
#else
    for (int tk = 0; tk < nk; ++tk) {
      const int kb = tk * 32;
      __syncthreads();
      *(v8h*)&lA[0][lofs]     = *(const v8h*)(pa0 + kb);
      *(v8h*)&lA[0][lofs + 8] = *(const v8h*)(pa0 + kb + 8);
      *(v8h*)&lB[0][lofs]     = *(const v8h*)(pb0 + kb);
      *(v8h*)&lB[0][lofs + 8] = *(const v8h*)(pb0 + kb + 8);
      __syncthreads();
      compute_tile(lA[0], lB[0]);
    }
#endif
  }

  // ---- Epilogue. C/D layout: VGPR r holds M = r + 8*half, lane N = lrow. ----
#pragma unroll
  for (int mi = 0; mi < 2; ++mi) {
#pragma unroll
    for (int ni = 0; ni < 4; ++ni) {
      const int gn = bn + wn + ni * 16 + lrow;
      float bval = 0.0f;
      if constexpr (MODE <= 2) bval = bias[gn];
#pragma unroll
      for (int r = 0; r < 8; ++r) {
        const int gm = bm + wm + mi * 16 + r + 8 * half;
        float v = acc[mi][ni][r];
        if constexpr (MODE == 0) {
          v = (v + bval) * scale;
          ((_Float16*)outp)[(size_t)gm * N + gn] = (_Float16)v;
        } else if constexpr (MODE == 1) {
          v = v + bval;
          ((_Float16*)outp)[(size_t)gm * N + gn] = (_Float16)v;
        } else if constexpr (MODE == 2) {
          v = v + bval;
          const int bb = gm >> 12;         // token row -> batch (4096 tokens/batch)
          const int nn = gm & 4095;
          ((_Float16*)outp)[((size_t)bb * N + gn) * 4096 + nn] = (_Float16)v;
        } else if constexpr (MODE == 3) {
          if (gn > gm) v = -30000.0f;      // causal mask (triu k=1): exp underflows to 0
          ((_Float16*)outp)[(size_t)gm * N + gn] = (_Float16)v;
        } else {
          ((float*)outp)[(size_t)gm * N + gn] = v;
        }
      }
    }
  }
}

// One block per row: max, sum(exp), normalize. N = 4096, 256 threads x 16 elems.
__global__ __launch_bounds__(256) void softmax_row(_Float16* __restrict__ P, int N)
{
  __shared__ float red[8];
  const int t = threadIdx.x;
  _Float16* row = P + (size_t)blockIdx.x * N;

  float v[16];
  {
    v8h a = *(const v8h*)(row + t * 16);
    v8h b = *(const v8h*)(row + t * 16 + 8);
#pragma unroll
    for (int i = 0; i < 8; ++i) { v[i] = (float)a[i]; v[8 + i] = (float)b[i]; }
  }

  float m = -3.4e38f;
#pragma unroll
  for (int i = 0; i < 16; ++i) m = fmaxf(m, v[i]);
#pragma unroll
  for (int off = 16; off; off >>= 1) m = fmaxf(m, __shfl_xor(m, off, 32));
  if ((t & 31) == 0) red[t >> 5] = m;
  __syncthreads();
  float mt = red[0];
#pragma unroll
  for (int i = 1; i < 8; ++i) mt = fmaxf(mt, red[i]);
  __syncthreads();

  float s = 0.0f;
#pragma unroll
  for (int i = 0; i < 16; ++i) { v[i] = __expf(v[i] - mt); s += v[i]; }
#pragma unroll
  for (int off = 16; off; off >>= 1) s += __shfl_xor(s, off, 32);
  if ((t & 31) == 0) red[t >> 5] = s;
  __syncthreads();
  float st = 0.0f;
#pragma unroll
  for (int i = 0; i < 8; ++i) st += red[i];
  const float inv = 1.0f / st;

  v8h a, b;
#pragma unroll
  for (int i = 0; i < 8; ++i) {
    a[i] = (_Float16)(v[i] * inv);
    b[i] = (_Float16)(v[8 + i] * inv);
  }
  *(v8h*)(row + t * 16)     = a;
  *(v8h*)(row + t * 16 + 8) = b;
}

extern "C" void kernel_launch(void* const* d_in, const int* in_sizes, int n_in,
                              void* d_out, int out_size, void* d_ws, size_t ws_size,
                              hipStream_t stream)
{
  const float* x  = (const float*)d_in[0];
  const float* Wq = (const float*)d_in[1];
  const float* bq = (const float*)d_in[2];
  const float* Wk = (const float*)d_in[3];
  const float* bk = (const float*)d_in[4];
  const float* Wv = (const float*)d_in[5];
  const float* bv = (const float*)d_in[6];
  float* out = (float*)d_out;

  const int D = 1024, S = 4096, B = 4, MT = B * S;   // MT = 16384 tokens

  // Workspace layout (128 MB total):
  //   [0,32MB)    Q   f16 [token][d]   (pre-scaled by 1/sqrt(D))
  //   [32,64MB)   K   f16 [token][d]
  //   [64,96MB)   V^T f16 [b][d][n]
  //   [96,128MB)  P   f16 [q][kv]      (reused per batch; batches stream-serialized)
  char* ws = (char*)d_ws;
  _Float16* Q  = (_Float16*)(ws);
  _Float16* Kb = (_Float16*)(ws + ((size_t)32 << 20));
  _Float16* Vt = (_Float16*)(ws + ((size_t)64 << 20));
  _Float16* P  = (_Float16*)(ws + ((size_t)96 << 20));

  dim3 blk(256);

  // QKV projections: y = x @ W^T + b  (W row-major [e][d] is exactly B^T layout [N][K])
  wmma_gemm<0><<<dim3(MT / 128, D / 128), blk, 0, stream>>>(x, Wq, Q,  bq, 0.03125f, MT, D, D, D, D);
  wmma_gemm<1><<<dim3(MT / 128, D / 128), blk, 0, stream>>>(x, Wk, Kb, bk, 1.0f,     MT, D, D, D, D);
  wmma_gemm<2><<<dim3(MT / 128, D / 128), blk, 0, stream>>>(x, Wv, Vt, bv, 1.0f,     MT, D, D, D, D);

  for (int b = 0; b < B; ++b) {
    const _Float16* Qb = Q  + (size_t)b * S * D;
    const _Float16* Kc = Kb + (size_t)b * S * D;
    const _Float16* Vb = Vt + (size_t)b * D * S;
    float* Ob = out + (size_t)b * S * D;
    // scores = Q K^T with causal mask, f16 (fully-masked 128x128 tiles skip all compute)
    wmma_gemm<3><<<dim3(S / 128, S / 128), blk, 0, stream>>>(Qb, Kc, P, nullptr, 1.0f, S, S, D, D, D);
    // row softmax in place
    softmax_row<<<dim3(S), blk, 0, stream>>>(P, S);
    // out = P @ V  (B^T = V^T [d][kv]; K clamped per row-block by causality)
    wmma_gemm<4><<<dim3(S / 128, D / 128), blk, 0, stream>>>(P, Vb, Ob, nullptr, 1.0f, S, D, S, S, S);
  }
}
